// Criterion_24489903522258
// MI455X (gfx1250) — matrix operand order repaired and checked by reading, not compiled, not run on hardware
//
#include <hip/hip_runtime.h>

#define BS 4096
#define DIM 512
#define POS_WF 2.0f
#define NEG_WF 40.0f
#define MARGIN 0.1f
#define POS_T 0.5f
#define NEG_T 0.5f

// LDS A-stripe row stride in ushorts: 512 + 24 pad -> dword stride 268 == 12 (mod 64)
// gives distinct 4-bank groups for the 16 b128 lane-loads, and 16B-aligned rows.
#define ASTRIDE 536

typedef __attribute__((ext_vector_type(16))) __bf16 v16bf;
typedef __attribute__((ext_vector_type(8)))  __bf16 v8bf;
typedef __attribute__((ext_vector_type(8)))  float  v8f;

__device__ __forceinline__ unsigned short f2bf_rne(float f) {
  unsigned u = __float_as_uint(f);
  unsigned r = u + 0x7FFFu + ((u >> 16) & 1u);
  return (unsigned short)(r >> 16);
}

// ---------------------------------------------------------------------------
// Kernel 0: split fp32 X into bf16 hi + bf16 lo (hi = rne(x), lo = rne(x-hi))
// ---------------------------------------------------------------------------
__global__ void __launch_bounds__(256) prep_split(const float* __restrict__ X,
                                                  unsigned short* __restrict__ Xhi,
                                                  unsigned short* __restrict__ Xlo) {
  int i = blockIdx.x * 256 + threadIdx.x;
  if (i < BS * DIM) {
    float v = X[i];
    unsigned short h = f2bf_rne(v);
    float hf = __uint_as_float(((unsigned)h) << 16);
    Xhi[i] = h;
    Xlo[i] = f2bf_rne(v - hf);
  }
}

// ---------------------------------------------------------------------------
// Split-bf16 WMMA core: one wave computes two 16x16 fp32 tiles (cols colA, colB)
// for the workgroup's 16-row stripe, accumulating over K = DIM.
//   sim ~= Ahi*Bhi + Ahi*Blo + Alo*Bhi   (fp32 accumulate)
// ---------------------------------------------------------------------------
__device__ __forceinline__ void gemm_two_tiles(
    const unsigned short* sAhi, const unsigned short* sAlo,
    const unsigned short* __restrict__ Xhi, const unsigned short* __restrict__ Xlo,
    int colA, int colB, int lane, v8f& acc0, v8f& acc1) {
  const int n = lane & 15;
  const int h = lane >> 4;
  const unsigned short* bAh = Xhi + (size_t)(colA + n) * DIM;
  const unsigned short* bAl = Xlo + (size_t)(colA + n) * DIM;
  const unsigned short* bBh = Xhi + (size_t)(colB + n) * DIM;
  const unsigned short* bBl = Xlo + (size_t)(colB + n) * DIM;
  const unsigned short* aRowH = sAhi + n * ASTRIDE;
  const unsigned short* aRowL = sAlo + n * ASTRIDE;

#pragma unroll 4
  for (int kc = 0; kc < DIM; kc += 32) {
    v8bf a0h = *(const v8bf*)(aRowH + kc + h * 16);
    v8bf a1h = *(const v8bf*)(aRowH + kc + 8 + h * 16);
    v8bf a0l = *(const v8bf*)(aRowL + kc + h * 16);
    v8bf a1l = *(const v8bf*)(aRowL + kc + 8 + h * 16);
    v16bf ah, al;
#pragma unroll
    for (int i = 0; i < 8; ++i) {
      ah[i] = a0h[i]; ah[8 + i] = a1h[i];
      al[i] = a0l[i]; al[8 + i] = a1l[i];
    }
    v16bf vbAh = *(const v16bf*)(bAh + kc + h * 16);
    v16bf vbAl = *(const v16bf*)(bAl + kc + h * 16);
    v16bf vbBh = *(const v16bf*)(bBh + kc + h * 16);
    v16bf vbBl = *(const v16bf*)(bBl + kc + h * 16);

    acc0 = __builtin_amdgcn_wmma_f32_16x16x32_bf16(false, ah, false, vbAh, (short)0, acc0, false, false);
    acc1 = __builtin_amdgcn_wmma_f32_16x16x32_bf16(false, ah, false, vbBh, (short)0, acc1, false, false);
    acc0 = __builtin_amdgcn_wmma_f32_16x16x32_bf16(false, ah, false, vbAl, (short)0, acc0, false, false);
    acc1 = __builtin_amdgcn_wmma_f32_16x16x32_bf16(false, ah, false, vbBl, (short)0, acc1, false, false);
    acc0 = __builtin_amdgcn_wmma_f32_16x16x32_bf16(false, al, false, vbAh, (short)0, acc0, false, false);
    acc1 = __builtin_amdgcn_wmma_f32_16x16x32_bf16(false, al, false, vbBh, (short)0, acc1, false, false);
  }
}

__device__ __forceinline__ void stage_A(const unsigned short* __restrict__ Xhi,
                                        const unsigned short* __restrict__ Xlo,
                                        unsigned short* sAhi, unsigned short* sAlo,
                                        int rowbase, int tid) {
  for (int i = tid; i < 16 * (DIM / 8); i += 256) {
    int r = i >> 6;
    int c = i & 63;
    const uint4* sh = (const uint4*)(Xhi + (size_t)(rowbase + r) * DIM) + c;
    const uint4* sl = (const uint4*)(Xlo + (size_t)(rowbase + r) * DIM) + c;
    *((uint4*)(sAhi + r * ASTRIDE) + c) = *sh;
    *((uint4*)(sAlo + r * ASTRIDE) + c) = *sl;
  }
}

// ---------------------------------------------------------------------------
// Pass 1: GEMM -> per-row bounds; optionally stores sim (fp32, row-major) for
// the L2-resident pass-2 path (64 MB << 192 MB L2).
//   pos_bound[i] = min_j sim[i,j] over same-class, j != i   (+inf if none)
//   neg_bound[i] = max_j sim[i,j] over diff-class           (-inf if none)
// ---------------------------------------------------------------------------
__global__ void __launch_bounds__(256) bounds_kernel(
    const unsigned short* __restrict__ Xhi, const unsigned short* __restrict__ Xlo,
    const int* __restrict__ labels,
    float* __restrict__ pos_bound, float* __restrict__ neg_bound,
    float* __restrict__ simOut) {
  __shared__ unsigned short sAhi[16 * ASTRIDE];
  __shared__ unsigned short sAlo[16 * ASTRIDE];
  __shared__ float sRed[8][16][2];

  const int tid = threadIdx.x;
  const int lane = tid & 31;
  const int wave = tid >> 5;
  const int rowbase = blockIdx.x * 16;
  const int n = lane & 15;
  const int h = lane >> 4;

  stage_A(Xhi, Xlo, sAhi, sAlo, rowbase, tid);
  __syncthreads();

  int labR[8];
#pragma unroll
  for (int e = 0; e < 8; ++e) labR[e] = labels[rowbase + 8 * h + e];

  float bminp[8], bmaxn[8];
#pragma unroll
  for (int e = 0; e < 8; ++e) { bminp[e] = __builtin_inff(); bmaxn[e] = -__builtin_inff(); }

  for (int it = 0; it < BS / 256; ++it) {
    const int colA = it * 256 + wave * 16;
    const int colB = colA + 128;
    const int labCA = labels[colA + n];
    const int labCB = labels[colB + n];
    v8f acc0 = {0.f, 0.f, 0.f, 0.f, 0.f, 0.f, 0.f, 0.f};
    v8f acc1 = {0.f, 0.f, 0.f, 0.f, 0.f, 0.f, 0.f, 0.f};
    gemm_two_tiles(sAhi, sAlo, Xhi, Xlo, colA, colB, lane, acc0, acc1);
#pragma unroll
    for (int e = 0; e < 8; ++e) {
      const int row = rowbase + 8 * h + e;
      {
        float s = acc0[e];
        bool same = (labR[e] == labCA);
        if (same && (row != (colA + n))) bminp[e] = fminf(bminp[e], s);
        if (!same) bmaxn[e] = fmaxf(bmaxn[e], s);
      }
      {
        float s = acc1[e];
        bool same = (labR[e] == labCB);
        if (same && (row != (colB + n))) bminp[e] = fminf(bminp[e], s);
        if (!same) bmaxn[e] = fmaxf(bmaxn[e], s);
      }
    }
    if (simOut != nullptr) {
#pragma unroll
      for (int e = 0; e < 8; ++e) {
        const size_t row = (size_t)(rowbase + 8 * h + e);
        simOut[row * BS + colA + n] = acc0[e];   // lanes n=0..15 -> coalesced 64B
        simOut[row * BS + colB + n] = acc1[e];
      }
    }
  }
#pragma unroll
  for (int e = 0; e < 8; ++e) {
    for (int o = 1; o < 16; o <<= 1) {
      bminp[e] = fminf(bminp[e], __shfl_xor(bminp[e], o, 32));
      bmaxn[e] = fmaxf(bmaxn[e], __shfl_xor(bmaxn[e], o, 32));
    }
  }
  if (n == 0) {
#pragma unroll
    for (int e = 0; e < 8; ++e) {
      sRed[wave][8 * h + e][0] = bminp[e];
      sRed[wave][8 * h + e][1] = bmaxn[e];
    }
  }
  __syncthreads();
  if (tid < 16) {
    float mp = __builtin_inff(), mx = -__builtin_inff();
    for (int w2 = 0; w2 < 8; ++w2) {
      mp = fminf(mp, sRed[w2][tid][0]);
      mx = fmaxf(mx, sRed[w2][tid][1]);
    }
    pos_bound[rowbase + tid] = mp;
    neg_bound[rowbase + tid] = mx;
  }
}

// ---------------------------------------------------------------------------
// Shared epilogue math for one element
// ---------------------------------------------------------------------------
__device__ __forceinline__ void lse_elem(float s, bool same, float pb, float nb,
                                         float& rminp, float& rmaxp, float& sump,
                                         float& rminn, float& rmaxn, float& sumn) {
  float wp = -POS_WF * (s - POS_T);
  float wn =  NEG_WF * (s - NEG_T);
  bool pm = same && (s - MARGIN < nb);
  bool nm = (!same) && (s + MARGIN > pb);
  float mp = pm ? wp : 0.f;
  float mn = nm ? wn : 0.f;
  rminp = fminf(rminp, mp); rmaxp = fmaxf(rmaxp, mp);
  rminn = fminf(rminn, mn); rmaxn = fmaxf(rmaxn, mn);
  sump += pm ? __expf(wp) : 0.f;   // wp <= 3  -> safe at ref=0
  sumn += nm ? __expf(wn) : 0.f;   // wn <= ~20 -> sum <= ~2e12, safe
}

// ---------------------------------------------------------------------------
// Pass 2a (big-ws path): streaming logsumexp over stored sim. One row per
// wave, float4 coalesced reads (L2-resident), labels staged in LDS.
// ---------------------------------------------------------------------------
__global__ void __launch_bounds__(256) lse_sim_kernel(
    const float* __restrict__ sim, const int* __restrict__ labels,
    const float* __restrict__ pos_bound, const float* __restrict__ neg_bound,
    float* __restrict__ posv, float* __restrict__ posnz,
    float* __restrict__ negv, float* __restrict__ negnz) {
  __shared__ int sLab[BS];
  const int tid = threadIdx.x;
  const int lane = tid & 31;
  const int wave = tid >> 5;
  for (int i = tid; i < BS / 4; i += 256) ((int4*)sLab)[i] = ((const int4*)labels)[i];
  __syncthreads();

  const int row = blockIdx.x * 8 + wave;
  const int labR = labels[row];
  const float pb = pos_bound[row];
  const float nb = neg_bound[row];
  const float4* srow = (const float4*)(sim + (size_t)row * BS);

  float rminp = __builtin_inff(), rmaxp = -__builtin_inff(), sump = 0.f;
  float rminn = __builtin_inff(), rmaxn = -__builtin_inff(), sumn = 0.f;

  for (int it = 0; it < BS / 128; ++it) {
    float4 v = srow[it * 32 + lane];
    int c0 = (it * 32 + lane) * 4;
    float sv[4] = {v.x, v.y, v.z, v.w};
#pragma unroll
    for (int q = 0; q < 4; ++q) {
      bool same = (labR == sLab[c0 + q]);
      lse_elem(sv[q], same, pb, nb, rminp, rmaxp, sump, rminn, rmaxn, sumn);
    }
  }
  for (int o = 1; o < 32; o <<= 1) {
    rminp = fminf(rminp, __shfl_xor(rminp, o, 32));
    rmaxp = fmaxf(rmaxp, __shfl_xor(rmaxp, o, 32));
    rminn = fminf(rminn, __shfl_xor(rminn, o, 32));
    rmaxn = fmaxf(rmaxn, __shfl_xor(rmaxn, o, 32));
    sump += __shfl_xor(sump, o, 32);
    sumn += __shfl_xor(sumn, o, 32);
  }
  if (lane == 0) {
    bool nzp = (rmaxp + rminp) != 0.f;
    bool nzn = (rmaxn + rminn) != 0.f;
    posv[row] = nzp ? logf(sump) : 0.f;  posnz[row] = nzp ? 1.f : 0.f;
    negv[row] = nzn ? logf(sumn) : 0.f;  negnz[row] = nzn ? 1.f : 0.f;
  }
}

// ---------------------------------------------------------------------------
// Pass 2b (fallback, small ws): recompute sim via WMMA and fuse the epilogue.
// ---------------------------------------------------------------------------
__global__ void __launch_bounds__(256) lse_kernel(
    const unsigned short* __restrict__ Xhi, const unsigned short* __restrict__ Xlo,
    const int* __restrict__ labels,
    const float* __restrict__ pos_bound, const float* __restrict__ neg_bound,
    float* __restrict__ posv, float* __restrict__ posnz,
    float* __restrict__ negv, float* __restrict__ negnz) {
  __shared__ unsigned short sAhi[16 * ASTRIDE];
  __shared__ unsigned short sAlo[16 * ASTRIDE];
  __shared__ float sRed[8][16][6];

  const int tid = threadIdx.x;
  const int lane = tid & 31;
  const int wave = tid >> 5;
  const int rowbase = blockIdx.x * 16;
  const int n = lane & 15;
  const int h = lane >> 4;

  stage_A(Xhi, Xlo, sAhi, sAlo, rowbase, tid);
  __syncthreads();

  int labR[8];
  float pbR[8], nbR[8];
#pragma unroll
  for (int e = 0; e < 8; ++e) {
    int r = rowbase + 8 * h + e;
    labR[e] = labels[r];
    pbR[e] = pos_bound[r];
    nbR[e] = neg_bound[r];
  }

  float rminp[8], rmaxp[8], sump[8], rminn[8], rmaxn[8], sumn[8];
#pragma unroll
  for (int e = 0; e < 8; ++e) {
    rminp[e] = __builtin_inff(); rmaxp[e] = -__builtin_inff(); sump[e] = 0.f;
    rminn[e] = __builtin_inff(); rmaxn[e] = -__builtin_inff(); sumn[e] = 0.f;
  }

  for (int it = 0; it < BS / 256; ++it) {
    const int colA = it * 256 + wave * 16;
    const int colB = colA + 128;
    const int labCA = labels[colA + n];
    const int labCB = labels[colB + n];
    v8f acc0 = {0.f, 0.f, 0.f, 0.f, 0.f, 0.f, 0.f, 0.f};
    v8f acc1 = {0.f, 0.f, 0.f, 0.f, 0.f, 0.f, 0.f, 0.f};
    gemm_two_tiles(sAhi, sAlo, Xhi, Xlo, colA, colB, lane, acc0, acc1);
#pragma unroll
    for (int e = 0; e < 8; ++e) {
      lse_elem(acc0[e], labR[e] == labCA, pbR[e], nbR[e],
               rminp[e], rmaxp[e], sump[e], rminn[e], rmaxn[e], sumn[e]);
      lse_elem(acc1[e], labR[e] == labCB, pbR[e], nbR[e],
               rminp[e], rmaxp[e], sump[e], rminn[e], rmaxn[e], sumn[e]);
    }
  }
#pragma unroll
  for (int e = 0; e < 8; ++e) {
    for (int o = 1; o < 16; o <<= 1) {
      rminp[e] = fminf(rminp[e], __shfl_xor(rminp[e], o, 32));
      rmaxp[e] = fmaxf(rmaxp[e], __shfl_xor(rmaxp[e], o, 32));
      rminn[e] = fminf(rminn[e], __shfl_xor(rminn[e], o, 32));
      rmaxn[e] = fmaxf(rmaxn[e], __shfl_xor(rmaxn[e], o, 32));
      sump[e] += __shfl_xor(sump[e], o, 32);
      sumn[e] += __shfl_xor(sumn[e], o, 32);
    }
  }
  if (n == 0) {
#pragma unroll
    for (int e = 0; e < 8; ++e) {
      int r = 8 * h + e;
      sRed[wave][r][0] = rminp[e]; sRed[wave][r][1] = rmaxp[e]; sRed[wave][r][2] = sump[e];
      sRed[wave][r][3] = rminn[e]; sRed[wave][r][4] = rmaxn[e]; sRed[wave][r][5] = sumn[e];
    }
  }
  __syncthreads();
  if (tid < 16) {
    float mnp = __builtin_inff(), mxp = -__builtin_inff(), sp = 0.f;
    float mnn = __builtin_inff(), mxn = -__builtin_inff(), sn = 0.f;
    for (int w2 = 0; w2 < 8; ++w2) {
      mnp = fminf(mnp, sRed[w2][tid][0]); mxp = fmaxf(mxp, sRed[w2][tid][1]); sp += sRed[w2][tid][2];
      mnn = fminf(mnn, sRed[w2][tid][3]); mxn = fmaxf(mxn, sRed[w2][tid][4]); sn += sRed[w2][tid][5];
    }
    bool nzp = (mxp + mnp) != 0.f;
    bool nzn = (mxn + mnn) != 0.f;
    int r = rowbase + tid;
    posv[r] = nzp ? logf(sp) : 0.f;  posnz[r] = nzp ? 1.f : 0.f;
    negv[r] = nzn ? logf(sn) : 0.f;  negnz[r] = nzn ? 1.f : 0.f;
  }
}

// ---------------------------------------------------------------------------
// Pass 3: masked mean of softplus(vals)/w for both paths -> scalar loss
// ---------------------------------------------------------------------------
__device__ __forceinline__ float softplusf(float x) {
  return fmaxf(x, 0.f) + log1pf(expf(-fabsf(x)));
}

__global__ void __launch_bounds__(256) final_kernel(
    const float* __restrict__ posv, const float* __restrict__ posnz,
    const float* __restrict__ negv, const float* __restrict__ negnz,
    float* __restrict__ out) {
  __shared__ float sP[256], sN[256], sCP[256], sCN[256];
  int t = threadIdx.x;
  float sp = 0.f, sn = 0.f, cp = 0.f, cn = 0.f;
  for (int i = t; i < BS; i += 256) {
    if (posnz[i] != 0.f) { sp += softplusf(posv[i]) / POS_WF; cp += 1.f; }
    if (negnz[i] != 0.f) { sn += softplusf(negv[i]) / NEG_WF; cn += 1.f; }
  }
  sP[t] = sp; sN[t] = sn; sCP[t] = cp; sCN[t] = cn;
  __syncthreads();
  for (int o = 128; o > 0; o >>= 1) {
    if (t < o) { sP[t] += sP[t + o]; sN[t] += sN[t + o]; sCP[t] += sCP[t + o]; sCN[t] += sCN[t + o]; }
    __syncthreads();
  }
  if (t == 0) {
    float lp = (sCP[0] > 0.f) ? sP[0] / fmaxf(sCP[0], 1.f) : (logf(2.f) / POS_WF);
    float ln = (sCN[0] > 0.f) ? sN[0] / fmaxf(sCN[0], 1.f) : (logf(2.f) / NEG_WF);
    out[0] = lp + ln;
  }
}

extern "C" void kernel_launch(void* const* d_in, const int* in_sizes, int n_in,
                              void* d_out, int out_size, void* d_ws, size_t ws_size,
                              hipStream_t stream) {
  (void)in_sizes; (void)n_in; (void)out_size;
  const float* X = (const float*)d_in[0];
  const int* labels = (const int*)d_in[1];
  float* out = (float*)d_out;

  char* w = (char*)d_ws;
  unsigned short* Xhi = (unsigned short*)w;                       // 4 MiB
  unsigned short* Xlo = Xhi + (size_t)BS * DIM;                   // 4 MiB
  float* pos_bound = (float*)(w + 2 * (size_t)BS * DIM * sizeof(unsigned short));
  float* neg_bound = pos_bound + BS;
  float* posv = neg_bound + BS;
  float* posnz = posv + BS;
  float* negv = posnz + BS;
  float* negnz = negv + BS;

  const size_t offSim = 2 * (size_t)BS * DIM * sizeof(unsigned short) +
                        8 * (size_t)BS * sizeof(float);           // 8 MiB + 128 KiB
  const size_t need = offSim + (size_t)BS * BS * sizeof(float);   // + 64 MiB
  const bool big = (ws_size >= need);
  float* sim = big ? (float*)(w + offSim) : nullptr;

  prep_split<<<(BS * DIM + 255) / 256, 256, 0, stream>>>(X, Xhi, Xlo);
  bounds_kernel<<<BS / 16, 256, 0, stream>>>(Xhi, Xlo, labels, pos_bound, neg_bound, sim);
  if (big) {
    lse_sim_kernel<<<BS / 8, 256, 0, stream>>>(sim, labels, pos_bound, neg_bound,
                                               posv, posnz, negv, negnz);
  } else {
    lse_kernel<<<BS / 16, 256, 0, stream>>>(Xhi, Xlo, labels, pos_bound, neg_bound,
                                            posv, posnz, negv, negnz);
  }
  final_kernel<<<1, 256, 0, stream>>>(posv, posnz, negv, negnz, out);
}
